// model_77472620085762
// MI455X (gfx1250) — compile-verified, matrix operand order checked by reading, not run on hardware
//
#include <hip/hip_runtime.h>
#include <hip/hip_bf16.h>

typedef __attribute__((ext_vector_type(16))) _Float16 v16h;
typedef __attribute__((ext_vector_type(8)))  _Float16 v8h;
typedef __attribute__((ext_vector_type(8)))  float    v8f;

#define HH   64
#define G3   192
#define TT   512
#define BB   4096
#define GROUPS 2              // row-groups per block (16 batch rows each)
#define WPG    2              // waves cooperating on one row-group
#define THREADS (GROUPS * WPG * 32)
#define ROWS_PER_BLOCK (GROUPS * 16)

#define WMMA_F16(A, B, C) \
  __builtin_amdgcn_wmma_f32_16x16x32_f16(false, (A), false, (B), (short)0, (C), false, false)

// ---- fast transcendentals: use native V_TANH_F32 / V_RCP_F32 / V_EXP_F32 ----
__device__ __forceinline__ float fast_rcp(float x) { return __builtin_amdgcn_rcpf(x); }
__device__ __forceinline__ float fast_exp2(float x) { return __builtin_amdgcn_exp2f(x); }

#if __has_builtin(__builtin_amdgcn_tanhf)
__device__ __forceinline__ float tanh_fast(float x) { return __builtin_amdgcn_tanhf(x); }
__device__ __forceinline__ float sigm(float x) {
  return __builtin_fmaf(__builtin_amdgcn_tanhf(0.5f * x), 0.5f, 0.5f);
}
#else
__device__ __forceinline__ float tanh_fast(float x) {
  float e = fast_exp2(x * -2.885390082f);          // e^(-2x)
  return (1.0f - e) * fast_rcp(1.0f + e);
}
__device__ __forceinline__ float sigm(float x) {
  return fast_rcp(1.0f + fast_exp2(x * -1.442695041f));
}
#endif

// A (row=m) / B (row=n) fragment from row-major [rows][64] f16 LDS:
// two contiguous 8-half runs -> two ds_load_b128.
__device__ __forceinline__ v16h load_frag(const _Float16* base, int row, int kc,
                                          int lhalf) {
  const _Float16* p = base + row * HH + kc * 32 + lhalf * 8;
  v8h lo = *(const v8h*)(p);
  v8h hi = *(const v8h*)(p + 16);
  return __builtin_shufflevector(lo, hi, 0, 1, 2, 3, 4, 5, 6, 7,
                                 8, 9, 10, 11, 12, 13, 14, 15);
}

__global__ __launch_bounds__(THREADS, 4) void gru2_fused_kernel(
    const float* __restrict__ x,
    const float* __restrict__ Wih0, const float* __restrict__ Whh0,
    const float* __restrict__ bih0, const float* __restrict__ bhh0,
    const float* __restrict__ Wih1, const float* __restrict__ Whh1,
    const float* __restrict__ bih1, const float* __restrict__ bhh1,
    const float* __restrict__ Wp, const float* __restrict__ bp,
    float* __restrict__ out) {
  __shared__ __align__(16) _Float16 sWhh0[G3 * HH];
  __shared__ __align__(16) _Float16 sWih1[G3 * HH];
  __shared__ __align__(16) _Float16 sWhh1[G3 * HH];
  __shared__ float sC0[G3];   // g<128: bih0+bhh0 ; g>=128: bih0 (n input-side)
  __shared__ float sC1[G3];
  __shared__ float sHn0[HH];  // bhh0[128+j]
  __shared__ float sHn1[HH];
  __shared__ float sWx[G3 * 2];
  __shared__ float sWp[2 * HH];
  __shared__ __align__(16) _Float16 h0buf[GROUPS][16 * HH];
  __shared__ __align__(16) _Float16 h1buf[GROUPS][16 * HH];

  const int tid = threadIdx.x;
  const int wave = tid >> 5;
  const int group = wave >> 1;      // which 16-row group
  const int par = wave & 1;         // jt-parity handled by this wave
  const int lane = tid & 31;
  const int lmod = lane & 15;
  const int lhalf = lane >> 4;

  for (int i = tid; i < G3 * HH; i += THREADS) {
    sWhh0[i] = (_Float16)Whh0[i];
    sWih1[i] = (_Float16)Wih1[i];
    sWhh1[i] = (_Float16)Whh1[i];
  }
  for (int i = tid; i < G3; i += THREADS) {
    sC0[i] = bih0[i] + (i < 128 ? bhh0[i] : 0.0f);
    sC1[i] = bih1[i] + (i < 128 ? bhh1[i] : 0.0f);
    sWx[2 * i] = Wih0[2 * i];
    sWx[2 * i + 1] = Wih0[2 * i + 1];
  }
  for (int i = tid; i < HH; i += THREADS) {
    sHn0[i] = bhh0[128 + i];
    sHn1[i] = bhh1[128 + i];
  }
  for (int i = tid; i < 2 * HH; i += THREADS) sWp[i] = Wp[i];
  {
    _Float16* hb0 = &h0buf[0][0];
    _Float16* hb1 = &h1buf[0][0];
    for (int i = tid; i < GROUPS * 16 * HH; i += THREADS) {
      hb0[i] = (_Float16)0.0f;
      hb1[i] = (_Float16)0.0f;
    }
  }
  __syncthreads();

  const int rowbase = blockIdx.x * ROWS_PER_BLOCK + group * 16;

  // h state in D layout, only for this wave's two jt units (jt = par + 2*ui):
  // h[ui][r] = h[m = r+8*lhalf, j = jt*16+lmod]
  float h0d[2][8], h1d[2][8];
#pragma unroll
  for (int ui = 0; ui < 2; ui++)
#pragma unroll
    for (int r = 0; r < 8; r++) { h0d[ui][r] = 0.0f; h1d[ui][r] = 0.0f; }

  const float* xbase = x + (size_t)rowbase * 2;

  for (int t = 0; t < TT; ++t) {
    const float* xr_ = xbase + (size_t)t * BB * 2;
    if (t + 4 < TT) __builtin_prefetch(xr_ + (size_t)4 * BB * 2, 0, 1);

    float x0m[8], x1m[8];
#pragma unroll
    for (int r = 0; r < 8; r++) {
      int m = r + 8 * lhalf;
      float2 v = *(const float2*)(xr_ + m * 2);
      x0m[r] = v.x; x1m[r] = v.y;
    }

    // old-h A fragments (both waves of the group read the same state)
    v16h a0[2], a1[2];
#pragma unroll
    for (int kc = 0; kc < 2; kc++) {
      a0[kc] = load_frag(h0buf[group], lmod, kc, lhalf);
      a1[kc] = load_frag(h1buf[group], lmod, kc, lhalf);
    }
    __syncthreads();  // B1: everyone has read old h0/h1 before it is overwritten

    // ---------------- layer 0 (this wave: jt = par, par+2) ----------------
#pragma unroll
    for (int ui = 0; ui < 2; ui++) {
      const int jt = par + 2 * ui;
      const int gb = jt * 16 + lmod;
      v8f accR = {}, accZ = {}, accN = {};
#pragma unroll
      for (int kc = 0; kc < 2; kc++) {
        v16h bR = load_frag(sWhh0, gb, kc, lhalf);
        v16h bZ = load_frag(sWhh0, 64 + gb, kc, lhalf);
        v16h bN = load_frag(sWhh0, 128 + gb, kc, lhalf);
        accR = WMMA_F16(a0[kc], bR, accR);
        accZ = WMMA_F16(a0[kc], bZ, accZ);
        accN = WMMA_F16(a0[kc], bN, accN);
      }
      const float c0r_ = sC0[gb], c0z_ = sC0[64 + gb];
      const float bi0n_ = sC0[128 + gb], bh0n_ = sHn0[gb];
      const float2 wr = *(const float2*)(sWx + 2 * gb);
      const float2 wz = *(const float2*)(sWx + 2 * (64 + gb));
      const float2 wn = *(const float2*)(sWx + 2 * (128 + gb));
#pragma unroll
      for (int r = 0; r < 8; r++) {
        float rg = sigm(accR[r] + wr.x * x0m[r] + wr.y * x1m[r] + c0r_);
        float zg = sigm(accZ[r] + wz.x * x0m[r] + wz.y * x1m[r] + c0z_);
        float ng = tanh_fast(wn.x * x0m[r] + wn.y * x1m[r] + bi0n_ +
                             rg * (accN[r] + bh0n_));
        float hn = (1.0f - zg) * ng + zg * h0d[ui][r];
        h0d[ui][r] = hn;
        h0buf[group][(r + 8 * lhalf) * HH + gb] = (_Float16)hn;
      }
    }
    __syncthreads();  // B2: full new h0 visible to both waves of the group

    v16h a0n[2];
#pragma unroll
    for (int kc = 0; kc < 2; kc++)
      a0n[kc] = load_frag(h0buf[group], lmod, kc, lhalf);

    // ---------------- layer 1 ----------------
#pragma unroll
    for (int ui = 0; ui < 2; ui++) {
      const int jt = par + 2 * ui;
      const int gb = jt * 16 + lmod;
      v8f aR = {}, aZ = {}, aXN = {}, aHN = {};
#pragma unroll
      for (int kc = 0; kc < 2; kc++) {
        v16h biR = load_frag(sWih1, gb, kc, lhalf);
        v16h bhR = load_frag(sWhh1, gb, kc, lhalf);
        aR = WMMA_F16(a0n[kc], biR, aR);
        aR = WMMA_F16(a1[kc], bhR, aR);
        v16h biZ = load_frag(sWih1, 64 + gb, kc, lhalf);
        v16h bhZ = load_frag(sWhh1, 64 + gb, kc, lhalf);
        aZ = WMMA_F16(a0n[kc], biZ, aZ);
        aZ = WMMA_F16(a1[kc], bhZ, aZ);
        v16h biN = load_frag(sWih1, 128 + gb, kc, lhalf);
        v16h bhN = load_frag(sWhh1, 128 + gb, kc, lhalf);
        aXN = WMMA_F16(a0n[kc], biN, aXN);  // n-gate: input/recurrent kept split
        aHN = WMMA_F16(a1[kc], bhN, aHN);
      }
      const float c1r_ = sC1[gb], c1z_ = sC1[64 + gb];
      const float bi1n_ = sC1[128 + gb], bh1n_ = sHn1[gb];
#pragma unroll
      for (int r = 0; r < 8; r++) {
        float rg = sigm(aR[r] + c1r_);
        float zg = sigm(aZ[r] + c1z_);
        float ng = tanh_fast(aXN[r] + bi1n_ + rg * (aHN[r] + bh1n_));
        float hn = (1.0f - zg) * ng + zg * h1d[ui][r];
        h1d[ui][r] = hn;
        h1buf[group][(r + 8 * lhalf) * HH + gb] = (_Float16)hn;
      }
    }
    __syncthreads();  // B3: new h1 (and h0) committed before next step's reads
  }

  // ----- projection: out[b] = Wp[0:64].h0f + Wp[64:128].h1f + bp -----
  if (par == 0 && lane < 16) {
    const int m = lane;
    float s = bp[0];
#pragma unroll 8
    for (int j = 0; j < HH; j++) {
      s += (float)h0buf[group][m * HH + j] * sWp[j];
      s += (float)h1buf[group][m * HH + j] * sWp[64 + j];
    }
    out[rowbase + m] = s;
  }
}

extern "C" void kernel_launch(void* const* d_in, const int* in_sizes, int n_in,
                              void* d_out, int out_size, void* d_ws, size_t ws_size,
                              hipStream_t stream) {
  const float* x    = (const float*)d_in[0];
  const float* Wih0 = (const float*)d_in[1];
  const float* Whh0 = (const float*)d_in[2];
  const float* bih0 = (const float*)d_in[3];
  const float* bhh0 = (const float*)d_in[4];
  const float* Wih1 = (const float*)d_in[5];
  const float* Whh1 = (const float*)d_in[6];
  const float* bih1 = (const float*)d_in[7];
  const float* bhh1 = (const float*)d_in[8];
  const float* Wp   = (const float*)d_in[9];
  const float* bp   = (const float*)d_in[10];
  float* out = (float*)d_out;

  dim3 grid(BB / ROWS_PER_BLOCK);   // 128 blocks
  dim3 block(THREADS);              // 4 waves: 2 row-groups x 2 coop waves
  gru2_fused_kernel<<<grid, block, 0, stream>>>(
      x, Wih0, Whh0, bih0, bhh0, Wih1, Whh1, bih1, bhh1, Wp, bp, out);
}
